// Metric_35545149342437
// MI455X (gfx1250) — compile-verified, matrix operand order checked by reading, not run on hardware
//
#include <hip/hip_runtime.h>
#include <math.h>

typedef __attribute__((ext_vector_type(16))) _Float16 v16h;
typedef __attribute__((ext_vector_type(8)))  _Float16 v8h;
typedef __attribute__((ext_vector_type(8)))  float    v8f;

// ---------------------------------------------------------------------------
// WMMA fragment helpers (wave32, ISA 7.12.2 layouts)
// A (16x32 f16): lane L holds row M=L&15; lanes<16 -> K {0..7,16..23},
//                lanes>=16 -> K {8..15,24..31}: two contiguous 8-half chunks.
// B (32x16 f16): pre-packed fragment-major -> one contiguous 32B load/lane.
// C/D (16x16 f32): VGPR r: lanes 0-15 -> M=r, lanes 16-31 -> M=r+8; N=lane&15.
// ---------------------------------------------------------------------------
__device__ __forceinline__ v16h frag_a_contig(const _Float16* rowp, int lane) {
    const _Float16* p = rowp + ((lane < 16) ? 0 : 8);
    v8h lo = *(const v8h*)p;
    v8h hi = *(const v8h*)(p + 16);
    return __builtin_shufflevector(lo, hi, 0, 1, 2, 3, 4, 5, 6, 7,
                                   8, 9, 10, 11, 12, 13, 14, 15);
}

#define WMMA_F16(af, bf, c) \
    __builtin_amdgcn_wmma_f32_16x16x32_f16(false, (af), false, (bf), (short)0, (c), false, false)

// ---------------------------------------------------------------------------
// Weight pre-pack: W (KxN row-major f32) -> fragment-major f16.
// ---------------------------------------------------------------------------
__global__ void pack_fc_w_kernel(const float* __restrict__ W, _Float16* __restrict__ out,
                                 int K, int N) {
    int total = (K >> 5) * (N >> 4) * 512;
    int idx = blockIdx.x * blockDim.x + threadIdx.x;
    if (idx >= total) return;
    int i = idx & 15;
    int lane = (idx >> 4) & 31;
    int t = idx >> 9;
    int ntiles = N >> 4;
    int nt = t % ntiles;
    int kc = t / ntiles;
    int base = (lane < 16) ? 0 : 8;
    int klocal = (i < 8) ? (base + i) : (16 + base + (i - 8));
    int k = kc * 32 + klocal;
    int n = nt * 16 + (lane & 15);
    out[idx] = (_Float16)W[(size_t)k * N + n];
}

// conv2-4 weights (64,64,3,3) -> frags [pos(9)][kc(2)][ot(4)][lane(32)][16]
__global__ void pack_conv_w_kernel(const float* __restrict__ w, _Float16* __restrict__ out) {
    int idx = blockIdx.x * blockDim.x + threadIdx.x;
    if (idx >= 36864) return;
    int i = idx & 15;
    int lane = (idx >> 4) & 31;
    int t = idx >> 9;                 // (pos*2+kc)*4+ot
    int ot = t & 3;
    int kc = (t >> 2) & 1;
    int pos = t >> 3;
    int base = (lane < 16) ? 0 : 8;
    int klocal = (i < 8) ? (base + i) : (16 + base + (i - 8));
    int ic = kc * 32 + klocal;
    int oc = ot * 16 + (lane & 15);
    out[idx] = (_Float16)w[((size_t)oc * 64 + ic) * 9 + pos];
}

// conv1 weights (64,3,3,3) -> frags [ot(4)][lane(32)][16], k = ky*9+kx*3+ic, pad>=27
__global__ void pack_conv1_w_kernel(const float* __restrict__ w, _Float16* __restrict__ out) {
    int idx = blockIdx.x * blockDim.x + threadIdx.x;
    if (idx >= 2048) return;
    int i = idx & 15;
    int lane = (idx >> 4) & 31;
    int ot = idx >> 9;
    int base = (lane < 16) ? 0 : 8;
    int klocal = (i < 8) ? (base + i) : (16 + base + (i - 8));
    int oc = ot * 16 + (lane & 15);
    float v = 0.f;
    if (klocal < 27) {
        int ky = klocal / 9;
        int r = klocal % 9;
        int kx = r / 3, ic = r % 3;
        v = w[((oc * 3 + ic) * 3 + ky) * 3 + kx];
    }
    out[idx] = (_Float16)v;
}

// images NCHW f32 (N,3,84,84) -> NHWC f16 (N,84,84,3)
__global__ void cvt_img_nhwc_kernel(const float* __restrict__ in, _Float16* __restrict__ out, int N) {
    int idx = blockIdx.x * blockDim.x + threadIdx.x;
    int total = N * 84 * 84 * 3;
    if (idx >= total) return;
    int c = idx % 3;
    int x = (idx / 3) % 84;
    int y = (idx / 252) % 84;
    int n = idx / 21168;
    out[idx] = (_Float16)in[(((size_t)n * 3 + c) * 84 + y) * 84 + x];
}

// ---------------------------------------------------------------------------
// WMMA conv1 (3ch->64ch) with in-register 2x2 maxpool.
// GEMM rows = pooled_pos*4 + quadrant; K = 27 taps padded to 32.
// img: f16 NHWC (N,84,84,3)  out: f32 NCHW (N,64,41,41)
// ---------------------------------------------------------------------------
__global__ __launch_bounds__(32) void wmma_conv1_kernel(
    const _Float16* __restrict__ img, const _Float16* __restrict__ wpack,
    const float* __restrict__ bias, float* __restrict__ out, int mtiles) {
    int mt = blockIdx.x % mtiles;
    int n  = blockIdx.x / mtiles;
    int lane = threadIdx.x & 31;
    int gm = mt * 16 + (lane & 15);
    int pp = gm >> 2;
    int quad = gm & 3;
    bool rowok = pp < 1681;
    if (!rowok) pp = 0;
    int y = 2 * (pp / 41) + (quad >> 1);
    int x = 2 * (pp % 41) + (quad & 1);
    const _Float16* p = img + (((size_t)n * 84 + y) * 84 + x) * 3;
    int base = (lane < 16) ? 0 : 8;
    v16h af = {};
    if (rowok) {
#pragma unroll
        for (int i = 0; i < 8; ++i) {
            int k0 = base + i;              // 0..15, always < 27
            af[i] = p[(k0 / 9) * 252 + (k0 % 9)];
            int k1 = 16 + base + i;         // 16..31
            af[i + 8] = (k1 < 27) ? p[(k1 / 9) * 252 + (k1 % 9)] : (_Float16)0.f;
        }
    }
    v8f c0 = WMMA_F16(af, *(const v16h*)(wpack + lane * 16),        (v8f){});
    v8f c1 = WMMA_F16(af, *(const v16h*)(wpack + 512 + lane * 16),  (v8f){});
    v8f c2 = WMMA_F16(af, *(const v16h*)(wpack + 1024 + lane * 16), (v8f){});
    v8f c3 = WMMA_F16(af, *(const v16h*)(wpack + 1536 + lane * 16), (v8f){});
    int nl = lane & 15;
#pragma unroll
    for (int ot = 0; ot < 4; ++ot) {
        v8f c = (ot == 0) ? c0 : (ot == 1) ? c1 : (ot == 2) ? c2 : c3;
        int oc = ot * 16 + nl;
        float bv = bias[oc];
#pragma unroll
        for (int g = 0; g < 2; ++g) {
            float mv = fmaxf(fmaxf(c[4 * g], c[4 * g + 1]),
                             fmaxf(c[4 * g + 2], c[4 * g + 3])) + bv;
            int ppo = mt * 4 + ((lane < 16) ? g : 2 + g);
            if (ppo < 1681) out[((size_t)n * 64 + oc) * 1681 + ppo] = mv;
        }
    }
}

// ---------------------------------------------------------------------------
// Implicit-GEMM WMMA conv3x3 (64->64), NHWC f16 input -> NCHW f32 output.
// A-fragment = two contiguous 16B channel-chunk loads per tap. One wave
// computes one 16-position tile x all 4 oc-tiles (A reuse).
// ---------------------------------------------------------------------------
__global__ __launch_bounds__(32) void wmma_conv_kernel(
    const _Float16* __restrict__ in, const _Float16* __restrict__ wpack,
    const float* __restrict__ bias, float* __restrict__ out,
    int N, int Hin, int Win, int Hc, int Wc, int mtiles) {
    int mt = blockIdx.x % mtiles;
    int n  = blockIdx.x / mtiles;
    int lane = threadIdx.x & 31;
    int HWc = Hc * Wc;
    int m = mt * 16 + (lane & 15);
    bool rowok = m < HWc;
    int y = rowok ? (m / Wc) : 0;
    int x = rowok ? (m % Wc) : 0;
    int base = (lane < 16) ? 0 : 8;
    const _Float16* inn = in + (size_t)n * Hin * Win * 64;

    v8f c0 = {}, c1 = {}, c2 = {}, c3 = {};
#pragma unroll
    for (int pos = 0; pos < 9; ++pos) {
        int ky = pos / 3, kx = pos % 3;
        const _Float16* p0 = inn + ((size_t)(y + ky) * Win + (x + kx)) * 64;
#pragma unroll
        for (int kc = 0; kc < 2; ++kc) {
            v16h af = {};
            if (rowok) {
                const _Float16* p = p0 + kc * 32 + base;
                v8h lo = *(const v8h*)p;
                v8h hi = *(const v8h*)(p + 16);
                af = __builtin_shufflevector(lo, hi, 0, 1, 2, 3, 4, 5, 6, 7,
                                             8, 9, 10, 11, 12, 13, 14, 15);
            }
            const _Float16* wb = wpack + (size_t)((pos * 2 + kc) * 4) * 512 + lane * 16;
            c0 = WMMA_F16(af, *(const v16h*)(wb),        c0);
            c1 = WMMA_F16(af, *(const v16h*)(wb + 512),  c1);
            c2 = WMMA_F16(af, *(const v16h*)(wb + 1024), c2);
            c3 = WMMA_F16(af, *(const v16h*)(wb + 1536), c3);
        }
    }
    int nl = lane & 15;
#pragma unroll
    for (int ot = 0; ot < 4; ++ot) {
        v8f c = (ot == 0) ? c0 : (ot == 1) ? c1 : (ot == 2) ? c2 : c3;
        int oc = ot * 16 + nl;
        float bv = bias[oc];
#pragma unroll
        for (int r = 0; r < 8; ++r) {
            int mm = mt * 16 + ((lane < 16) ? r : r + 8);
            if (mm < HWc) out[((size_t)n * 64 + oc) * HWc + mm] = c[r] + bv;
        }
    }
}

// 2x2 maxpool stride 2 (NCHW f32)
__global__ void maxpool2_kernel(const float* __restrict__ in, float* __restrict__ out,
                                int N, int Hc, int Wc, int Ho, int Wo) {
    int idx = blockIdx.x * blockDim.x + threadIdx.x;
    if (idx >= N * 64 * Ho * Wo) return;
    int ox = idx % Wo;
    int oy = (idx / Wo) % Ho;
    int c  = (idx / (Wo * Ho)) & 63;
    int n  = idx / (Wo * Ho * 64);
    const float* p = in + ((size_t)(n * 64 + c) * Hc + oy * 2) * Wc + ox * 2;
    out[idx] = fmaxf(fmaxf(p[0], p[1]), fmaxf(p[Wc], p[Wc + 1]));
}

// BatchNorm stats: block per channel. stats[c]=mean, stats[64+c]=rstd
__global__ void bn_stats_kernel(const float* __restrict__ x, float* __restrict__ stats,
                                int N, int HW) {
    int c = blockIdx.x;
    int tid = threadIdx.x;
    float s = 0.f, s2 = 0.f;
    int total = N * HW;
    for (int t = tid; t < total; t += blockDim.x) {
        int n = t / HW, r = t % HW;
        float v = x[((size_t)n * 64 + c) * HW + r];
        s += v; s2 += v * v;
    }
    __shared__ float rs[256], rs2[256];
    rs[tid] = s; rs2[tid] = s2;
    __syncthreads();
    for (int o = 128; o > 0; o >>= 1) {
        if (tid < o) { rs[tid] += rs[tid + o]; rs2[tid] += rs2[tid + o]; }
        __syncthreads();
    }
    if (tid == 0) {
        float cnt = (float)total;
        float m = rs[0] / cnt;
        float var = rs2[0] / cnt - m * m;
        stats[c] = m;
        stats[64 + c] = rsqrtf(var + 1e-5f);
    }
}

// BN + ReLU: NCHW f32 -> NHWC f16 (transpose on write, coalesced stores)
__global__ void bn_relu16_nhwc_kernel(const float* __restrict__ x, const float* __restrict__ stats,
                                      const float* __restrict__ g, const float* __restrict__ bb,
                                      _Float16* __restrict__ out, int N, int HW) {
    int idx = blockIdx.x * blockDim.x + threadIdx.x;
    int total = N * HW * 64;
    if (idx >= total) return;
    int c = idx & 63;
    int r = (idx >> 6) % HW;
    int n = idx / (HW * 64);
    float v = (x[((size_t)n * 64 + c) * HW + r] - stats[c]) * stats[64 + c] * g[c] + bb[c];
    out[idx] = (_Float16)fmaxf(v, 0.f);
}

// avgpool 5x5 stride 5: NHWC f16 (N,15,15,64) -> f32 (N,64,9)
__global__ void avgpool16_kernel(const _Float16* __restrict__ x, float* __restrict__ rep, int N) {
    int idx = blockIdx.x * blockDim.x + threadIdx.x;
    if (idx >= N * 64 * 9) return;
    int p = idx % 9;
    int c = (idx / 9) & 63;
    int n = idx / 576;
    int i = p / 3, j = p % 3;
    const _Float16* bp = x + (((size_t)n * 15 + i * 5) * 15 + j * 5) * 64 + c;
    float s = 0.f;
#pragma unroll
    for (int yy = 0; yy < 5; ++yy)
#pragma unroll
        for (int xx = 0; xx < 5; ++xx) s += (float)bp[(yy * 15 + xx) * 64];
    rep[idx] = s * (1.f / 25.f);
}

// ---------------------------------------------------------------------------
// Projection: out[ns,p,k] = sum_{c<64} rep[ns,c,p]*W[c,k] + (i/3)W[64,k] + (j/3)W[65,k]
// ---------------------------------------------------------------------------
__global__ void proj_kernel(const float* __restrict__ rep, const float* __restrict__ W,
                            float* __restrict__ out) {
    int p = blockIdx.x % 9;
    int ns = blockIdx.x / 9;
    int k = threadIdx.x;
    const float* rp = rep + (size_t)ns * 576 + p;
    float acc = 0.f;
    for (int c = 0; c < 64; ++c) acc += rp[c * 9] * W[c * 256 + k];
    acc += (float)(p / 3) * (1.f / 3.f) * W[64 * 256 + k]
         + (float)(p % 3) * (1.f / 3.f) * W[65 * 256 + k];
    out[(size_t)blockIdx.x * 256 + k] = acc;
}

// ---------------------------------------------------------------------------
// Fused relation g-MLP: one block = 48 of 81 (ps,pq) rows of one (b,q,s)
// through three 256x256 WMMA layers, LDS-resident; deterministic reduction.
// ---------------------------------------------------------------------------
__global__ __launch_bounds__(256) void relation_g_kernel(
    const float* __restrict__ Abuf, const float* __restrict__ Bqbuf,
    const float* __restrict__ gb1,
    const _Float16* __restrict__ W2p, const float* __restrict__ gb2,
    const _Float16* __restrict__ W3p, const float* __restrict__ gb3,
    const _Float16* __restrict__ W4p, const float* __restrict__ gb4,
    float* __restrict__ x_f) {
    int hlf = blockIdx.x & 1;
    int bqs = blockIdx.x >> 1;
    int s = bqs % 5;
    int q = (bqs / 5) % 15;
    int b = bqs / 75;

    __shared__ _Float16 hbuf0[48 * 256];
    __shared__ _Float16 hbuf1[48 * 256];

    int tid = threadIdx.x;
    int lane = tid & 31;
    int wave = tid >> 5;

    for (int lr = 0; lr < 48; ++lr) {
        int gr = hlf * 48 + lr;
        float v = 0.f;
        if (gr < 81) {
            int ps = gr / 9, pq = gr % 9;
            v = Abuf[((size_t)(b * 5 + s) * 9 + ps) * 256 + tid]
              + Bqbuf[((size_t)(b * 15 + q) * 9 + pq) * 256 + tid]
              + gb1[tid];
            v = fmaxf(v, 0.f);
        }
        hbuf0[lr * 256 + tid] = (_Float16)v;
    }
    __syncthreads();

    for (int layer = 0; layer < 3; ++layer) {
        const _Float16* Wp = (layer == 0) ? W2p : (layer == 1) ? W3p : W4p;
        const float* bias = (layer == 0) ? gb2 : (layer == 1) ? gb3 : gb4;
        const _Float16* cur = (layer & 1) ? hbuf1 : hbuf0;
        _Float16* nxt = (layer & 1) ? hbuf0 : hbuf1;
        for (int i = 0; i < 6; ++i) {
            int t = wave * 6 + i;
            int mt = t >> 4, nt = t & 15;
            const _Float16* arow = cur + mt * 16 * 256 + (size_t)(lane & 15) * 256;
            v8f c = {};
#pragma unroll
            for (int kc = 0; kc < 8; ++kc) {
                v16h af = frag_a_contig(arow + kc * 32, lane);
                v16h bf = *(const v16h*)(Wp + (size_t)(kc * 16 + nt) * 512 + lane * 16);
                c = WMMA_F16(af, bf, c);
            }
            int n = nt * 16 + (lane & 15);
            float bn_ = bias[n];
#pragma unroll
            for (int r = 0; r < 8; ++r) {
                int ml = mt * 16 + ((lane < 16) ? r : r + 8);
                float v = fmaxf(c[r] + bn_, 0.f);
                if (layer == 2 && (hlf * 48 + ml) >= 81) v = 0.f;
                nxt[ml * 256 + n] = (_Float16)v;
            }
        }
        __syncthreads();
    }

    float ssum = 0.f;
    for (int lr = 0; lr < 48; ++lr) ssum += (float)hbuf1[lr * 256 + tid];
    atomicAdd(&x_f[(size_t)bqs * 256 + tid], ssum);
}

// ---------------------------------------------------------------------------
// Generic WMMA FC: out = [relu](A(MxK,f16) @ Wpacked + bias). 1 wave / tile.
// ---------------------------------------------------------------------------
__global__ __launch_bounds__(32) void wmma_fc_kernel(
    const _Float16* __restrict__ A, const _Float16* __restrict__ Wp,
    const float* __restrict__ bias, _Float16* __restrict__ out16,
    float* __restrict__ out32, int M, int N, int K, int dorelu) {
    int lane = threadIdx.x & 31;
    int ntiles = N >> 4;
    int mt = blockIdx.x / ntiles;
    int nt = blockIdx.x % ntiles;
    int row = mt * 16 + (lane & 15);
    bool rowok = row < M;
    const _Float16* arow = A + (size_t)row * K;
    v8f c = {};
    for (int kc = 0; kc < (K >> 5); ++kc) {
        v16h af = {};
        if (rowok) af = frag_a_contig(arow + kc * 32, lane);
        v16h bf = *(const v16h*)(Wp + (size_t)(kc * ntiles + nt) * 512 + lane * 16);
        c = WMMA_F16(af, bf, c);
    }
    int n = nt * 16 + (lane & 15);
    float bn_ = bias[n];
#pragma unroll
    for (int r = 0; r < 8; ++r) {
        int m = mt * 16 + ((lane < 16) ? r : r + 8);
        if (m < M) {
            float v = c[r] + bn_;
            if (dorelu) v = fmaxf(v, 0.f);
            if (out16) out16[(size_t)m * N + n] = (_Float16)v;
            if (out32) out32[(size_t)m * N + n] = v;
        }
    }
}

// ---------------------------------------------------------------------------
// Scoring + margin loss per (b,q), deterministic
// ---------------------------------------------------------------------------
__global__ __launch_bounds__(32) void loss_kernel(const float* __restrict__ obuf,
                                                  const int* __restrict__ sup_y,
                                                  const int* __restrict__ qry_y,
                                                  float* __restrict__ lossbuf) {
    int bq = blockIdx.x;
    int b = bq / 15;
    int lane = threadIdx.x;
    __shared__ float sc[5];
    for (int s = 0; s < 5; ++s) {
        const float* op = obuf + (size_t)(bq * 5 + s) * 64;
        float p = 0.f;
        for (int k = lane; k < 64; k += 32) p += op[k] * op[k];
#pragma unroll
        for (int o = 16; o > 0; o >>= 1) p += __shfl_xor(p, o, 32);
        if (lane == 0) sc[s] = sqrtf(p);
    }
    __syncthreads();
    if (lane == 0) {
        float nn2 = 0.f;
        for (int s = 0; s < 5; ++s) nn2 += sc[s] * sc[s];
        float nrm = sqrtf(nn2);
        float scale = (nrm * nrm / (1.f + nrm * nrm)) / nrm;
        int qy = qry_y[bq];
        float sp[5];
        float sap = 0.f;
        for (int s = 0; s < 5; ++s) {
            sp[s] = sc[s] * scale;
            if (sup_y[b * 5 + s] == qy) sap += sp[s];
        }
        float loss = 0.f;
        for (int s = 0; s < 5; ++s)
            if (sup_y[b * 5 + s] != qy) loss += fmaxf(sp[s] - sap + 0.2f, 0.f);
        lossbuf[bq] = loss;
    }
}

__global__ void final_reduce_kernel(const float* __restrict__ part, float* __restrict__ out, int n) {
    __shared__ float sh[128];
    float s = 0.f;
    for (int i = threadIdx.x; i < n; i += 128) s += part[i];
    sh[threadIdx.x] = s;
    __syncthreads();
    for (int o = 64; o > 0; o >>= 1) {
        if (threadIdx.x < o) sh[threadIdx.x] += sh[threadIdx.x + o];
        __syncthreads();
    }
    if (threadIdx.x == 0) out[0] = sh[0];
}

__global__ void cvt_f16_kernel(const float* __restrict__ in, _Float16* __restrict__ out, int n) {
    int i = blockIdx.x * blockDim.x + threadIdx.x;
    if (i < n) out[i] = (_Float16)in[i];
}

__global__ void zero_kernel(float* __restrict__ p, int n) {
    int i = blockIdx.x * blockDim.x + threadIdx.x;
    if (i < n) p[i] = 0.f;
}

// ---------------------------------------------------------------------------
static inline int du(long n, int blk) { return (int)((n + blk - 1) / blk); }

extern "C" void kernel_launch(void* const* d_in, const int* in_sizes, int n_in,
                              void* d_out, int out_size, void* d_ws, size_t ws_size,
                              hipStream_t stream) {
    (void)in_sizes; (void)n_in; (void)out_size; (void)ws_size;

    const float* sup_x = (const float*)d_in[0];
    const int*   sup_y = (const int*)d_in[1];
    const float* qry_x = (const float*)d_in[2];
    const int*   qry_y = (const int*)d_in[3];
    const float* cw1 = (const float*)d_in[4],  *cb1 = (const float*)d_in[5];
    const float* bg1 = (const float*)d_in[6],  *bb1 = (const float*)d_in[7];
    const float* cw2 = (const float*)d_in[8],  *cb2 = (const float*)d_in[9];
    const float* bg2 = (const float*)d_in[10], *bb2 = (const float*)d_in[11];
    const float* cw3 = (const float*)d_in[12], *cb3 = (const float*)d_in[13];
    const float* bg3 = (const float*)d_in[14], *bb3 = (const float*)d_in[15];
    const float* cw4 = (const float*)d_in[16], *cb4 = (const float*)d_in[17];
    const float* bg4 = (const float*)d_in[18], *bb4 = (const float*)d_in[19];
    const float* gW1 = (const float*)d_in[20], *gb1 = (const float*)d_in[21];
    const float* gW2 = (const float*)d_in[22], *gb2 = (const float*)d_in[23];
    const float* gW3 = (const float*)d_in[24], *gb3 = (const float*)d_in[25];
    const float* gW4 = (const float*)d_in[26], *gb4 = (const float*)d_in[27];
    const float* fW1 = (const float*)d_in[28], *fb1 = (const float*)d_in[29];
    const float* fW2 = (const float*)d_in[30], *fb2 = (const float*)d_in[31];
    const float* fW3 = (const float*)d_in[32], *fb3 = (const float*)d_in[33];
    const float* fW4 = (const float*)d_in[34], *fb4 = (const float*)d_in[35];

    // ---- workspace layout ----
    char* ws = (char*)d_ws;
    size_t off = 0;
    auto alloc = [&](size_t bytes) -> char* {
        char* p = ws + off;
        off += (bytes + 255) & ~(size_t)255;
        return p;
    };
    float* bufA     = (float*)alloc((size_t)12910080 * 4);    // f32 stage buffer
    float* bufB     = (float*)alloc((size_t)2772480 * 4);     // f32 pooled buffer
    _Float16* in16  = (_Float16*)alloc((size_t)12910080 * 2); // f16 NHWC activations
    _Float16* img16 = (_Float16*)alloc((size_t)2540160 * 2);  // f16 NHWC images
    float* supRep   = (float*)alloc((size_t)23040 * 4);
    float* qryRep   = (float*)alloc((size_t)69120 * 4);
    float* Abuf     = (float*)alloc((size_t)92160 * 4);
    float* Bqbuf    = (float*)alloc((size_t)276480 * 4);
    float* x_f      = (float*)alloc((size_t)153600 * 4);
    _Float16* xf16  = (_Float16*)alloc((size_t)153600 * 2);
    _Float16* h1h   = (_Float16*)alloc((size_t)153600 * 2);
    _Float16* h2h   = (_Float16*)alloc((size_t)153600 * 2);
    _Float16* h3h   = (_Float16*)alloc((size_t)76800 * 2);
    float* obuf     = (float*)alloc((size_t)38400 * 4);
    float* stats    = (float*)alloc(128 * 4);
    float* lossbuf  = (float*)alloc(120 * 4);
    _Float16* wp1   = (_Float16*)alloc((size_t)2048 * 2);     // conv1 weight frags
    _Float16* wp2   = (_Float16*)alloc((size_t)36864 * 2);
    _Float16* wp3   = (_Float16*)alloc((size_t)36864 * 2);
    _Float16* wp4   = (_Float16*)alloc((size_t)36864 * 2);
    _Float16* gW2p  = (_Float16*)alloc((size_t)65536 * 2);
    _Float16* gW3p  = (_Float16*)alloc((size_t)65536 * 2);
    _Float16* gW4p  = (_Float16*)alloc((size_t)65536 * 2);
    _Float16* fW1p  = (_Float16*)alloc((size_t)65536 * 2);
    _Float16* fW2p  = (_Float16*)alloc((size_t)65536 * 2);
    _Float16* fW3p  = (_Float16*)alloc((size_t)32768 * 2);
    _Float16* fW4p  = (_Float16*)alloc((size_t)8192 * 2);

    // ---- weight packing + init ----
    pack_conv1_w_kernel<<<du(2048, 256), 256, 0, stream>>>(cw1, wp1);
    pack_conv_w_kernel<<<du(36864, 256), 256, 0, stream>>>(cw2, wp2);
    pack_conv_w_kernel<<<du(36864, 256), 256, 0, stream>>>(cw3, wp3);
    pack_conv_w_kernel<<<du(36864, 256), 256, 0, stream>>>(cw4, wp4);
    pack_fc_w_kernel<<<du(65536, 256), 256, 0, stream>>>(gW2, gW2p, 256, 256);
    pack_fc_w_kernel<<<du(65536, 256), 256, 0, stream>>>(gW3, gW3p, 256, 256);
    pack_fc_w_kernel<<<du(65536, 256), 256, 0, stream>>>(gW4, gW4p, 256, 256);
    pack_fc_w_kernel<<<du(65536, 256), 256, 0, stream>>>(fW1, fW1p, 256, 256);
    pack_fc_w_kernel<<<du(65536, 256), 256, 0, stream>>>(fW2, fW2p, 256, 256);
    pack_fc_w_kernel<<<du(32768, 256), 256, 0, stream>>>(fW3, fW3p, 256, 128);
    pack_fc_w_kernel<<<du(8192, 256), 256, 0, stream>>>(fW4, fW4p, 128, 64);
    zero_kernel<<<du(153600, 256), 256, 0, stream>>>(x_f, 153600);

    // ---- representation network (BN stats per sub-batch, as in reference) ----
    auto repnet = [&](const float* x, int N, float* rep) {
        // layer 1: WMMA conv (K=27->32) + in-register maxpool -> f32 NCHW
        cvt_img_nhwc_kernel<<<du((long)N * 21168, 256), 256, 0, stream>>>(x, img16, N);
        int mt1 = (6724 + 15) / 16; // 421
        wmma_conv1_kernel<<<N * mt1, 32, 0, stream>>>(img16, wp1, cb1, bufA, mt1);
        int t1 = N * 64 * 1681;
        bn_stats_kernel<<<64, 256, 0, stream>>>(bufA, stats, N, 1681);
        bn_relu16_nhwc_kernel<<<du(t1, 256), 256, 0, stream>>>(bufA, stats, bg1, bb1, in16, N, 1681);
        // layer 2: WMMA conv 41->39, pool ->19
        int mt2 = (1521 + 15) / 16; // 96
        wmma_conv_kernel<<<N * mt2, 32, 0, stream>>>(in16, wp2, cb2, bufA, N, 41, 41, 39, 39, mt2);
        int t2 = N * 64 * 361;
        maxpool2_kernel<<<du(t2, 256), 256, 0, stream>>>(bufA, bufB, N, 39, 39, 19, 19);
        bn_stats_kernel<<<64, 256, 0, stream>>>(bufB, stats, N, 361);
        bn_relu16_nhwc_kernel<<<du(t2, 256), 256, 0, stream>>>(bufB, stats, bg2, bb2, in16, N, 361);
        // layer 3: WMMA conv 19->17
        int mt3 = (289 + 15) / 16; // 19
        wmma_conv_kernel<<<N * mt3, 32, 0, stream>>>(in16, wp3, cb3, bufA, N, 19, 19, 17, 17, mt3);
        int t3 = N * 64 * 289;
        bn_stats_kernel<<<64, 256, 0, stream>>>(bufA, stats, N, 289);
        bn_relu16_nhwc_kernel<<<du(t3, 256), 256, 0, stream>>>(bufA, stats, bg3, bb3, in16, N, 289);
        // layer 4: WMMA conv 17->15
        int mt4 = (225 + 15) / 16; // 15
        wmma_conv_kernel<<<N * mt4, 32, 0, stream>>>(in16, wp4, cb4, bufB, N, 17, 17, 15, 15, mt4);
        int t4 = N * 64 * 225;
        bn_stats_kernel<<<64, 256, 0, stream>>>(bufB, stats, N, 225);
        bn_relu16_nhwc_kernel<<<du(t4, 256), 256, 0, stream>>>(bufB, stats, bg4, bb4, in16, N, 225);
        // avgpool 5x5 -> (N,64,9)
        avgpool16_kernel<<<du(N * 64 * 9, 256), 256, 0, stream>>>(in16, rep, N);
    };
    repnet(sup_x, 40, supRep);
    repnet(qry_x, 120, qryRep);

    // ---- projections ----
    proj_kernel<<<40 * 9, 256, 0, stream>>>(supRep, gW1, Abuf);
    proj_kernel<<<120 * 9, 256, 0, stream>>>(qryRep, gW1 + 66 * 256, Bqbuf);

    // ---- fused relation g-MLP (WMMA, LDS-resident h) ----
    relation_g_kernel<<<1200, 256, 0, stream>>>(Abuf, Bqbuf, gb1,
                                                gW2p, gb2, gW3p, gb3, gW4p, gb4, x_f);

    // ---- f-MLP (WMMA) ----
    cvt_f16_kernel<<<du(153600, 256), 256, 0, stream>>>(x_f, xf16, 153600);
    wmma_fc_kernel<<<38 * 16, 32, 0, stream>>>(xf16, fW1p, fb1, h1h, nullptr, 600, 256, 256, 1);
    wmma_fc_kernel<<<38 * 16, 32, 0, stream>>>(h1h, fW2p, fb2, h2h, nullptr, 600, 256, 256, 1);
    wmma_fc_kernel<<<38 * 8, 32, 0, stream>>>(h2h, fW3p, fb3, h3h, nullptr, 600, 128, 256, 1);
    wmma_fc_kernel<<<38 * 4, 32, 0, stream>>>(h3h, fW4p, fb4, nullptr, obuf, 600, 64, 128, 0);

    // ---- scoring + margin loss ----
    loss_kernel<<<120, 32, 0, stream>>>(obuf, sup_y, qry_y, lossbuf);
    final_reduce_kernel<<<1, 128, 0, stream>>>(lossbuf, (float*)d_out, 120);
}
